// EGNN_67138928771579
// MI455X (gfx1250) — matrix operand order repaired and checked by reading, not compiled, not run on hardware
//
#include <hip/hip_runtime.h>
#include <hip/hip_bf16.h>

// CDNA5 / gfx1250 EGNN implementation.
// Edge + node MLPs run on v_wmma_f32_16x16x32_f16; activations staged in LDS
// (f16, 16B-aligned row strides -> ds_load_b128 fragment loads); weights are
// pre-transposed to f16 [n][k] so B fragments are contiguous global_load_b128
// (warmed into WGP$/L2 via global_prefetch_b8); segment sums via
// global_atomic_add_f32.

typedef __attribute__((ext_vector_type(16))) _Float16 v16h;
typedef __attribute__((ext_vector_type(8)))  _Float16 v8h;
typedef __attribute__((ext_vector_type(8)))  float    v8f;

#define HD 64   // hidden size

// ---- WMMA fragment helpers (wave32, 16x16x32 f16) --------------------------
// A (16xK tile, row-major f16, lda halves): lane<16 row=lane, elems 0..7 =
// K k0..k0+7, elems 8..15 = K k0+16..k0+23; lane>=16 same row, +8 shift.
__device__ __forceinline__ v16h load_a_frag(const _Float16* base, int lda, int k0, int lane) {
  const int r  = lane & 15;
  const int hi = lane >> 4;
  const _Float16* p = base + r * lda + k0 + hi * 8;
  v8h lo = *(const v8h*)(p);
  v8h hh = *(const v8h*)(p + 16);
  return __builtin_shufflevector(lo, hh, 0, 1, 2, 3, 4, 5, 6, 7,
                                         8, 9, 10, 11, 12, 13, 14, 15);
}

// B (Kx16 tile) from transposed weights Wt[n][k] (ldb halves): lane<16 holds
// column n0+lane, elems 0..7 = K k0..k0+7, elems 8..15 = K k0+16..k0+23;
// lane>=16 holds column n0+lane-16, same with +8 shift.
__device__ __forceinline__ v16h load_b_frag(const _Float16* wt, int ldb, int n0, int k0, int lane) {
  const int r  = lane & 15;
  const int hi = lane >> 4;
  const _Float16* p = wt + (size_t)(n0 + r) * ldb + k0 + hi * 8;
  v8h a = *(const v8h*)(p);
  v8h b = *(const v8h*)(p + 16);
  return __builtin_shufflevector(a, b, 0, 1, 2, 3, 4, 5, 6, 7,
                                       8, 9, 10, 11, 12, 13, 14, 15);
}

__device__ __forceinline__ float siluf(float v) { return v / (1.0f + __expf(-v)); }

// Warm weight bytes into WGP$/L2: emits global_prefetch_b8 (gfx1250).
__device__ __forceinline__ void prefetch_bytes(const void* p, int bytes, int tid, int nthr) {
  const char* c = (const char*)p;
  for (int off = tid * 128; off < bytes; off += nthr * 128)
    __builtin_prefetch(c + off, 0, 0);
}

// ---- weight prep: f32 [K][64] -> f16 transposed [64][Kpad] (zero padded) ---
__global__ void wprep(const float* __restrict__ src, _Float16* __restrict__ dst,
                      int K, int Nout, int Kpad) {
  int i = blockIdx.x * blockDim.x + threadIdx.x;
  if (i >= Nout * Kpad) return;
  int n = i / Kpad, k = i % Kpad;
  float v = (k < K) ? src[(size_t)k * Nout + n] : 0.0f;
  dst[(size_t)n * Kpad + k] = (_Float16)v;
}

// ---- edge kernel: per-wave 16-edge tile ------------------------------------
__global__ __launch_bounds__(128) void egnn_edge(
    const float* __restrict__ hin, const float* __restrict__ xin,
    const float* __restrict__ efeat,
    const int* __restrict__ esrc, const int* __restrict__ edst,
    const _Float16* __restrict__ ew1t, const float* __restrict__ eb1,
    const _Float16* __restrict__ ew2t, const float* __restrict__ eb2,
    const _Float16* __restrict__ cw1t, const float* __restrict__ cb1,
    const float* __restrict__ cw2,
    float* __restrict__ hagg, float* __restrict__ xagg, int nE)
{
  __shared__ __align__(16) _Float16 sF [4][16 * 160];  // edge-MLP input rows
  __shared__ __align__(16) _Float16 sMa[4][16 * 64];   // m1
  __shared__ __align__(16) _Float16 sMb[4][16 * 64];   // m2 (messages)
  __shared__ float sXdn[4][16][3];
  __shared__ float sCoef[4][16];

  const int wave  = threadIdx.x >> 5;
  const int lane  = threadIdx.x & 31;
  const int tile0 = (blockIdx.x * 4 + wave) * 16;
  _Float16* fT = &sF[wave][0];

  // Warm the (heavily reused, L2-resident) weight tiles.
  prefetch_bytes(ew1t, 64 * 160 * 2, threadIdx.x, 128);
  prefetch_bytes(ew2t, 64 * 64 * 2, threadIdx.x, 128);
  prefetch_bytes(cw1t, 64 * 64 * 2, threadIdx.x, 128);

  // Phase A: gather h[src], h[dst], radial, ef -> f16 rows [16 x 160]
  for (int j = 0; j < 16; ++j) {
    const int e = tile0 + j;
    const bool valid = (e < nE);
    const int s = valid ? esrc[e] : 0;
    const int d = valid ? edst[e] : 0;
    float2 hs = make_float2(0.f, 0.f), hd = make_float2(0.f, 0.f);
    if (valid) {
      hs = *(const float2*)(hin + (size_t)s * HD + lane * 2);
      hd = *(const float2*)(hin + (size_t)d * HD + lane * 2);
    }
    fT[j * 160 + lane * 2 + 0]      = (_Float16)hs.x;
    fT[j * 160 + lane * 2 + 1]      = (_Float16)hs.y;
    fT[j * 160 + 64 + lane * 2 + 0] = (_Float16)hd.x;
    fT[j * 160 + 64 + lane * 2 + 1] = (_Float16)hd.y;
    if (lane == 0) {
      float dx = 0.f, dy = 0.f, dz = 0.f;
      if (valid) {
        dx = xin[(size_t)s * 3 + 0] - xin[(size_t)d * 3 + 0];
        dy = xin[(size_t)s * 3 + 1] - xin[(size_t)d * 3 + 1];
        dz = xin[(size_t)s * 3 + 2] - xin[(size_t)d * 3 + 2];
      }
      float rad = dx * dx + dy * dy + dz * dz;
      float inv = 1.0f / (sqrtf(rad) + 1e-30f);
      sXdn[wave][j][0] = dx * inv;
      sXdn[wave][j][1] = dy * inv;
      sXdn[wave][j][2] = dz * inv;
      fT[j * 160 + 128] = (_Float16)rad;
    }
    if (lane >= 1 && lane <= 4) {
      float v = valid ? efeat[(size_t)e * 4 + (lane - 1)] : 0.f;
      fT[j * 160 + 128 + lane] = (_Float16)v;
    }
    if (lane >= 5) fT[j * 160 + 128 + lane] = (_Float16)0.f;  // pad 133..159
  }
  __syncthreads();

  // Phase B: m1 = silu(f @ ew1 + eb1)   (5 K-tiles x 4 N-tiles = 20 WMMA)
#pragma unroll
  for (int nt = 0; nt < 4; ++nt) {
    v8f acc = {};
#pragma unroll
    for (int kt = 0; kt < 5; ++kt) {
      v16h a = load_a_frag(fT, 160, kt * 32, lane);
      v16h b = load_b_frag(ew1t, 160, nt * 16, kt * 32, lane);
      acc = __builtin_amdgcn_wmma_f32_16x16x32_f16(false, a, false, b, (short)0, acc, false, false);
    }
    const int c = lane & 15, hi = lane >> 4;
    const float bias = eb1[nt * 16 + c];
#pragma unroll
    for (int v = 0; v < 8; ++v)
      sMa[wave][(v + 8 * hi) * 64 + nt * 16 + c] = (_Float16)siluf(acc[v] + bias);
  }
  __syncthreads();

  // Phase C: m2 = silu(m1 @ ew2 + eb2)  (8 WMMA)
#pragma unroll
  for (int nt = 0; nt < 4; ++nt) {
    v8f acc = {};
#pragma unroll
    for (int kt = 0; kt < 2; ++kt) {
      v16h a = load_a_frag(&sMa[wave][0], 64, kt * 32, lane);
      v16h b = load_b_frag(ew2t, 64, nt * 16, kt * 32, lane);
      acc = __builtin_amdgcn_wmma_f32_16x16x32_f16(false, a, false, b, (short)0, acc, false, false);
    }
    const int c = lane & 15, hi = lane >> 4;
    const float bias = eb2[nt * 16 + c];
#pragma unroll
    for (int v = 0; v < 8; ++v)
      sMb[wave][(v + 8 * hi) * 64 + nt * 16 + c] = (_Float16)siluf(acc[v] + bias);
  }
  __syncthreads();

  // Phase D: t = silu(m2 @ cw1 + cb1) -> reuse fT region as [16 x 64] (8 WMMA)
#pragma unroll
  for (int nt = 0; nt < 4; ++nt) {
    v8f acc = {};
#pragma unroll
    for (int kt = 0; kt < 2; ++kt) {
      v16h a = load_a_frag(&sMb[wave][0], 64, kt * 32, lane);
      v16h b = load_b_frag(cw1t, 64, nt * 16, kt * 32, lane);
      acc = __builtin_amdgcn_wmma_f32_16x16x32_f16(false, a, false, b, (short)0, acc, false, false);
    }
    const int c = lane & 15, hi = lane >> 4;
    const float bias = cb1[nt * 16 + c];
#pragma unroll
    for (int v = 0; v < 8; ++v)
      fT[(v + 8 * hi) * 64 + nt * 16 + c] = (_Float16)siluf(acc[v] + bias);
  }
  __syncthreads();

  // Phase E: coef = t @ cw2 (64 -> 1), one edge per lane 0..15
  if (lane < 16) {
    float a = 0.f;
    for (int k = 0; k < 64; ++k) a += (float)fT[lane * 64 + k] * cw2[k];
    sCoef[wave][lane] = a;
  }
  __syncthreads();

  // Phase F: segment-sum scatter (h_agg += m2 ; x_agg += coef * x_diff_norm)
  for (int j = 0; j < 16; ++j) {
    const int e = tile0 + j;
    if (e < nE) {
      const int d = edst[e];
      float m0 = (float)sMb[wave][j * 64 + lane * 2 + 0];
      float m1 = (float)sMb[wave][j * 64 + lane * 2 + 1];
      atomicAdd(&hagg[(size_t)d * HD + lane * 2 + 0], m0);
      atomicAdd(&hagg[(size_t)d * HD + lane * 2 + 1], m1);
      if (lane < 3)
        atomicAdd(&xagg[(size_t)d * 3 + lane], sCoef[wave][j] * sXdn[wave][j][lane]);
    }
  }
}

// ---- node kernel: h_out = silu(concat[h,h_agg] @ nw1 + nb1) @ nw2 + nb2 ----
__global__ __launch_bounds__(128) void egnn_node(
    const float* __restrict__ hin, const float* __restrict__ hagg,
    const _Float16* __restrict__ nw1t, const float* __restrict__ nb1,
    const _Float16* __restrict__ nw2t, const float* __restrict__ nb2,
    float* __restrict__ hout, float* __restrict__ osum, int nN, int doRelu)
{
  __shared__ __align__(16) _Float16 sC[4][16 * 128];
  __shared__ __align__(16) _Float16 sU[4][16 * 64];
  const int wave  = threadIdx.x >> 5;
  const int lane  = threadIdx.x & 31;
  const int tile0 = (blockIdx.x * 4 + wave) * 16;
  _Float16* cT = &sC[wave][0];

  prefetch_bytes(nw1t, 64 * 128 * 2, threadIdx.x, 128);
  prefetch_bytes(nw2t, 64 * 64 * 2, threadIdx.x, 128);

  for (int j = 0; j < 16; ++j) {
    const int n = tile0 + j;
    float2 a = make_float2(0.f, 0.f), g = make_float2(0.f, 0.f);
    if (n < nN) {
      a = *(const float2*)(hin  + (size_t)n * HD + lane * 2);
      g = *(const float2*)(hagg + (size_t)n * HD + lane * 2);
    }
    cT[j * 128 + lane * 2 + 0]      = (_Float16)a.x;
    cT[j * 128 + lane * 2 + 1]      = (_Float16)a.y;
    cT[j * 128 + 64 + lane * 2 + 0] = (_Float16)g.x;
    cT[j * 128 + 64 + lane * 2 + 1] = (_Float16)g.y;
  }
  __syncthreads();

#pragma unroll
  for (int nt = 0; nt < 4; ++nt) {           // u = silu(c @ nw1 + nb1): 16 WMMA
    v8f acc = {};
#pragma unroll
    for (int kt = 0; kt < 4; ++kt) {
      v16h a = load_a_frag(cT, 128, kt * 32, lane);
      v16h b = load_b_frag(nw1t, 128, nt * 16, kt * 32, lane);
      acc = __builtin_amdgcn_wmma_f32_16x16x32_f16(false, a, false, b, (short)0, acc, false, false);
    }
    const int c = lane & 15, hi = lane >> 4;
    const float bias = nb1[nt * 16 + c];
#pragma unroll
    for (int v = 0; v < 8; ++v)
      sU[wave][(v + 8 * hi) * 64 + nt * 16 + c] = (_Float16)siluf(acc[v] + bias);
  }
  __syncthreads();

#pragma unroll
  for (int nt = 0; nt < 4; ++nt) {           // h_out = u @ nw2 + nb2: 8 WMMA
    v8f acc = {};
#pragma unroll
    for (int kt = 0; kt < 2; ++kt) {
      v16h a = load_a_frag(&sU[wave][0], 64, kt * 32, lane);
      v16h b = load_b_frag(nw2t, 64, nt * 16, kt * 32, lane);
      acc = __builtin_amdgcn_wmma_f32_16x16x32_f16(false, a, false, b, (short)0, acc, false, false);
    }
    const int c = lane & 15, hi = lane >> 4;
    const float bias = nb2[nt * 16 + c];
    if (hout) {
#pragma unroll
      for (int v = 0; v < 8; ++v) {
        const int n = tile0 + v + 8 * hi;
        if (n < nN) {
          float val = acc[v] + bias;
          if (doRelu) val = fmaxf(val, 0.f);
          hout[(size_t)n * HD + nt * 16 + c] = val;
        }
      }
    }
    if (osum) {                              // final dgl.sum_nodes -> [1,64]
      float part = 0.f;
#pragma unroll
      for (int v = 0; v < 8; ++v) {
        const int n = tile0 + v + 8 * hi;
        if (n < nN) part += acc[v] + bias;
      }
      atomicAdd(&osum[nt * 16 + c], part);
    }
  }
}

// ---- x update: x1 = relu(x + x_agg) ----------------------------------------
__global__ void xupdate(const float* __restrict__ x, const float* __restrict__ xagg,
                        float* __restrict__ xout, int count) {
  int i = blockIdx.x * blockDim.x + threadIdx.x;
  if (i < count) xout[i] = fmaxf(x[i] + xagg[i], 0.f);
}

extern "C" void kernel_launch(void* const* d_in, const int* in_sizes, int n_in,
                              void* d_out, int out_size, void* d_ws, size_t ws_size,
                              hipStream_t stream) {
  (void)n_in; (void)out_size; (void)ws_size;
  const float* node_feat = (const float*)d_in[0];
  const float* coord     = (const float*)d_in[1];
  const float* efeat     = (const float*)d_in[2];
  const int*   esrc      = (const int*)d_in[3];
  const int*   edst      = (const int*)d_in[4];
  const int N = in_sizes[0] / HD;
  const int E = in_sizes[3];

  auto al = [](size_t x) { return (x + 255) & ~(size_t)255; };
  char* p = (char*)d_ws;
  float* h1   = (float*)p; p += al(sizeof(float) * (size_t)N * HD);
  float* hagg = (float*)p; p += al(sizeof(float) * (size_t)N * HD);
  float* x1   = (float*)p; p += al(sizeof(float) * (size_t)N * 3);
  float* xagg = (float*)p; p += al(sizeof(float) * (size_t)N * 3);
  _Float16* wbase = (_Float16*)p;
  const size_t O_EW1 = 0;
  const size_t O_EW2 = O_EW1 + (size_t)64 * 160;
  const size_t O_CW1 = O_EW2 + (size_t)64 * 64;
  const size_t O_NW1 = O_CW1 + (size_t)64 * 64;
  const size_t O_NW2 = O_NW1 + (size_t)64 * 128;
  const size_t LSTR  = O_NW2 + (size_t)64 * 64;

  // Pre-transpose all f32 weights to f16 [n][Kpad]
  for (int l = 0; l < 2; ++l) {
    const int b = 5 + 11 * l;
    _Float16* w = wbase + l * LSTR;
    struct { int idx; size_t off; int K, Kpad; } jobs[5] = {
      { b + 0, O_EW1, 133, 160 }, { b + 2, O_EW2, 64, 64 },
      { b + 4, O_CW1, 64, 64 },   { b + 7, O_NW1, 128, 128 },
      { b + 9, O_NW2, 64, 64 } };
    for (int t = 0; t < 5; ++t) {
      int total = 64 * jobs[t].Kpad;
      wprep<<<(total + 255) / 256, 256, 0, stream>>>(
          (const float*)d_in[jobs[t].idx], w + jobs[t].off, jobs[t].K, 64, jobs[t].Kpad);
    }
  }

  const int egrid = (E + 63) / 64;   // 4 waves x 16 edges per block
  const int ngrid = (N + 63) / 64;   // 4 waves x 16 nodes per block

  for (int l = 0; l < 2; ++l) {
    const int b = 5 + 11 * l;
    _Float16* w = wbase + l * LSTR;
    hipMemsetAsync(hagg, 0, sizeof(float) * (size_t)N * HD, stream);
    hipMemsetAsync(xagg, 0, sizeof(float) * (size_t)N * 3, stream);
    const float* hcur = (l == 0) ? node_feat : h1;
    const float* xcur = (l == 0) ? coord : x1;
    egnn_edge<<<egrid, 128, 0, stream>>>(
        hcur, xcur, efeat, esrc, edst,
        w + O_EW1, (const float*)d_in[b + 1],
        w + O_EW2, (const float*)d_in[b + 3],
        w + O_CW1, (const float*)d_in[b + 5],
        (const float*)d_in[b + 6],
        hagg, xagg, E);
    if (l == 0) {
      egnn_node<<<ngrid, 128, 0, stream>>>(
          hcur, hagg, w + O_NW1, (const float*)d_in[b + 8],
          w + O_NW2, (const float*)d_in[b + 10], h1, nullptr, N, 1);
      xupdate<<<(N * 3 + 255) / 256, 256, 0, stream>>>(coord, xagg, x1, N * 3);
    } else {
      hipMemsetAsync(d_out, 0, sizeof(float) * 64, stream);
      egnn_node<<<ngrid, 128, 0, stream>>>(
          hcur, hagg, w + O_NW1, (const float*)d_in[b + 8],
          w + O_NW2, (const float*)d_in[b + 10], nullptr, (float*)d_out, N, 0);
    }
  }
}
